// SubMconv3dBlock_55387898249948
// MI455X (gfx1250) — matrix operand order repaired and checked by reading, not compiled, not run on hardware
//
#include <hip/hip_runtime.h>

#define GRIDX 160
#define C_IN 64
#define C_OUT 96
#define TAPS 27
#define B_TAP_HALVES 6144   // 12 tiles * 512 bf16 = 12KB per tap

typedef __bf16 bf16x4  __attribute__((ext_vector_type(4)));
typedef __bf16 bf16x8  __attribute__((ext_vector_type(8)));
typedef __bf16 bf16x16 __attribute__((ext_vector_type(16)));
typedef float  f32x8   __attribute__((ext_vector_type(8)));
typedef unsigned int u32x4 __attribute__((ext_vector_type(4)));
typedef int i32x4 __attribute__((ext_vector_type(4)));
typedef int i32x8 __attribute__((ext_vector_type(8)));

// ---------------- grid build ----------------
__global__ void init_grid_k(int* __restrict__ g, int n) {
  int i = blockIdx.x * blockDim.x + threadIdx.x;
  if (i < n) g[i] = -1;
}

__global__ void scatter_k(const int* __restrict__ coords, int* __restrict__ g, int n) {
  int i = blockIdx.x * blockDim.x + threadIdx.x;
  if (i < n) {
    int z = coords[3 * i + 0], y = coords[3 * i + 1], x = coords[3 * i + 2];
    atomicMax(&g[(z * GRIDX + y) * GRIDX + x], i);  // deterministic: max index wins
  }
}

// ---------------- fp32 -> bf16 features ----------------
__global__ void cvt_bf16_k(const float* __restrict__ in, __bf16* __restrict__ out, int n4) {
  int i = blockIdx.x * blockDim.x + threadIdx.x;
  if (i < n4) {
    float4 f = reinterpret_cast<const float4*>(in)[i];
    bf16x4 o;
    o[0] = (__bf16)f.x; o[1] = (__bf16)f.y; o[2] = (__bf16)f.z; o[3] = (__bf16)f.w;
    reinterpret_cast<bf16x4*>(out)[i] = o;
  }
}

// ---------------- pack weights into WMMA B-matrix lane layout ----------------
// packed[tile][lane][j], tile = kidx*12 + kstep*6 + nt; B 32x16 bf16 layout:
// lanes 0-15 hold K=0..15 (j=K), lanes 16-31 hold K=16..31; N = lane&15.
__global__ void pack_w_k(const float* __restrict__ w, __bf16* __restrict__ pw) {
  int tid = blockIdx.x * blockDim.x + threadIdx.x;
  if (tid >= TAPS * 12 * 512) return;
  int tile = tid >> 9, within = tid & 511;
  int lane = within >> 4, j = within & 15;
  int kidx = tile / 12, rem = tile % 12;
  int kstep = rem / 6, nt = rem % 6;
  int khigh = lane >> 4, n = lane & 15;
  int K = kstep * 32 + khigh * 16 + j;
  int co = nt * 16 + n;
  pw[tid] = (__bf16)w[(kidx * C_IN + K) * C_OUT + co];
}

// ---------------- TDM: one-row tile of 6144 bf16 -> LDS ----------------
#if __has_builtin(__builtin_amdgcn_tensor_load_to_lds)
#define USE_TDM 1
__device__ __forceinline__ void tdm_load_b_tap(const __bf16* gsrc, unsigned lds_off) {
  unsigned long long ga = (unsigned long long)(uintptr_t)gsrc;
  u32x4 g0;
  g0[0] = 1u;                                             // count=1, user mode
  g0[1] = lds_off;                                        // lds_addr
  g0[2] = (unsigned)ga;                                   // global_addr[31:0]  (bits 95:64)
  g0[3] = (unsigned)((ga >> 32) & 0x01FFFFFFu) | 0x80000000u;  // addr[56:32], type=2
  i32x8 g1;
  g1[0] = 0x00010000;          // workgroup_mask=0, data_size=1 (2 bytes)
  g1[1] = (int)(B_TAP_HALVES << 16);   // tensor_dim0[15:0] in [63:48]
  g1[2] = 0x00010000;          // tensor_dim0 hi=0, tensor_dim1=1
  g1[3] = (int)(B_TAP_HALVES << 16);   // tile_dim0 in [127:112]
  g1[4] = 1;                   // tile_dim1=1, tile_dim2=0
  g1[5] = B_TAP_HALVES;        // tensor_dim0_stride[31:0]
  g1[6] = 0;
  g1[7] = 0;
  i32x4 gz; gz[0] = 0; gz[1] = 0; gz[2] = 0; gz[3] = 0;
#if defined(__clang_major__) && __clang_major__ >= 23
  i32x8 gz8; gz8[0]=0; gz8[1]=0; gz8[2]=0; gz8[3]=0; gz8[4]=0; gz8[5]=0; gz8[6]=0; gz8[7]=0;
  __builtin_amdgcn_tensor_load_to_lds(g0, g1, gz, gz, gz8, 0);
#else
  __builtin_amdgcn_tensor_load_to_lds(g0, g1, gz, gz, 0);
#endif
}
#else
#define USE_TDM 0
#endif

// ---------------- LayerNorm(96) + gamma/beta + ReLU epilogue for one 16x96 tile ----
__device__ __forceinline__ void ln_relu_store(
    f32x8* acc, const float* bv, const float* gv, const float* btv,
    float* __restrict__ out, int rowBase, int n, int N)
{
#pragma unroll
  for (int nt = 0; nt < 6; ++nt)
#pragma unroll
    for (int r = 0; r < 8; ++r) acc[nt][r] += bv[nt];

  float meanv[8], rstdv[8];
#pragma unroll
  for (int r = 0; r < 8; ++r) {
    float s1 = 0.f, s2 = 0.f;
#pragma unroll
    for (int nt = 0; nt < 6; ++nt) { float v = acc[nt][r]; s1 += v; s2 += v * v; }
#pragma unroll
    for (int d = 1; d < 16; d <<= 1) {
      s1 += __shfl_xor(s1, d, 32);
      s2 += __shfl_xor(s2, d, 32);
    }
    float m   = s1 * (1.0f / 96.0f);
    float var = s2 * (1.0f / 96.0f) - m * m;
    meanv[r] = m;
    rstdv[r] = rsqrtf(var + 1e-5f);
  }
#pragma unroll
  for (int nt = 0; nt < 6; ++nt)
#pragma unroll
    for (int r = 0; r < 8; ++r) {
      int row = rowBase + r;
      if (row < N) {
        float v = (acc[nt][r] - meanv[r]) * rstdv[r] * gv[nt] + btv[nt];
        out[(size_t)row * C_OUT + nt * 16 + n] = fmaxf(v, 0.0f);
      }
    }
}

// ---------------- fused subm-conv + LayerNorm + ReLU ----------------
// 256 threads = 8 waves; each wave owns two 16-row M-tiles (block covers 256 rows).
// Per-tap B block (12KB) staged in LDS (double buffered) by the Tensor Data Mover;
// B tiles double-buffered in registers inside the nt loop.
__global__ __launch_bounds__(256) void subm_conv_ln_relu_k(
    const __bf16* __restrict__ featB,
    const int* __restrict__ coords,
    const __bf16* __restrict__ packW,
    const int* __restrict__ vgrid,
    const float* __restrict__ bias_p,
    const float* __restrict__ gamma_p,
    const float* __restrict__ beta_p,
    float* __restrict__ out, int N)
{
  __shared__ __attribute__((aligned(32))) __bf16 sB[2][B_TAP_HALVES];

  const int lane   = threadIdx.x & 31;
  const int wave   = threadIdx.x >> 5;
  const int mbase  = (blockIdx.x * 8 + wave) * 32;   // 2 M-tiles per wave
  const int n      = lane & 15;
  const int laneHi = lane >> 4;
  const int row0   = mbase + n;
  const int row1   = mbase + 16 + n;

  const bool v0 = row0 < N, v1 = row1 < N;
  const int cz0 = v0 ? coords[3 * row0 + 0] : -10000;
  const int cy0 = v0 ? coords[3 * row0 + 1] : -10000;
  const int cx0 = v0 ? coords[3 * row0 + 2] : -10000;
  const int cz1 = v1 ? coords[3 * row1 + 0] : -10000;
  const int cy1 = v1 ? coords[3 * row1 + 1] : -10000;
  const int cx1 = v1 ? coords[3 * row1 + 2] : -10000;

  f32x8 acc0[6], acc1[6];
  f32x8 zf = {};
#pragma unroll
  for (int t = 0; t < 6; ++t) { acc0[t] = zf; acc1[t] = zf; }

  // Prologue: fetch tap 0 B block into buffer 0.
#if USE_TDM
  if (wave == 0)
    tdm_load_b_tap(packW, (unsigned)(uintptr_t)&sB[0][0]);
#else
  {
    const uint4* s = (const uint4*)packW;
    uint4* d = (uint4*)&sB[0][0];
    for (int t = threadIdx.x; t < B_TAP_HALVES / 8; t += 256) d[t] = s[t];
  }
#endif

  for (int kidx = 0; kidx < TAPS; ++kidx) {
    const int cur = kidx & 1;
#if USE_TDM
    if (wave == 0) __builtin_amdgcn_s_wait_tensorcnt(0);
#endif
    __syncthreads();  // B[cur] visible; previous readers of B[cur^1] are done
    if (kidx + 1 < TAPS) {
      const __bf16* src = packW + (size_t)(kidx + 1) * B_TAP_HALVES;
#if USE_TDM
      if (wave == 0)
        tdm_load_b_tap(src, (unsigned)(uintptr_t)&sB[cur ^ 1][0]);
#else
      const uint4* s = (const uint4*)src;
      uint4* d = (uint4*)&sB[cur ^ 1][0];
      for (int t = threadIdx.x; t < B_TAP_HALVES / 8; t += 256) d[t] = s[t];
#endif
    }

    // Kick off the first B-tile pair from LDS so it overlaps the A gathers below.
    const __bf16* sb = &sB[cur][lane * 16];
    bf16x16 bc0 = *reinterpret_cast<const bf16x16*>(sb);              // nt=0, kstep 0
    bf16x16 bc1 = *reinterpret_cast<const bf16x16*>(sb + 6 * 512);    // nt=0, kstep 1

    const int dz = kidx / 9 - 1;
    const int dy = (kidx / 3) % 3 - 1;
    const int dx = kidx % 3 - 1;

    int nidx0 = -1, nidx1 = -1;
    {
      int z = cz0 + dz, y = cy0 + dy, x = cx0 + dx;
      if ((unsigned)z < GRIDX && (unsigned)y < GRIDX && (unsigned)x < GRIDX)
        nidx0 = vgrid[(z * GRIDX + y) * GRIDX + x];
      z = cz1 + dz; y = cy1 + dy; x = cx1 + dx;
      if ((unsigned)z < GRIDX && (unsigned)y < GRIDX && (unsigned)x < GRIDX)
        nidx1 = vgrid[(z * GRIDX + y) * GRIDX + x];
    }

    // A tiles 16x32 bf16: lanes 0-15 K=j, lanes 16-31 K=16+j (matches packed B).
    bf16x16 a00 = {}, a01 = {}, a10 = {}, a11 = {};
    if (nidx0 >= 0) {
      const __bf16* p = featB + (long long)nidx0 * C_IN + laneHi * 16;
      bf16x8 l0 = *reinterpret_cast<const bf16x8*>(p);
      bf16x8 h0 = *reinterpret_cast<const bf16x8*>(p + 8);
      bf16x8 l1 = *reinterpret_cast<const bf16x8*>(p + 32);
      bf16x8 h1 = *reinterpret_cast<const bf16x8*>(p + 40);
      a00 = __builtin_shufflevector(l0, h0, 0,1,2,3,4,5,6,7,8,9,10,11,12,13,14,15);
      a01 = __builtin_shufflevector(l1, h1, 0,1,2,3,4,5,6,7,8,9,10,11,12,13,14,15);
    }
    if (nidx1 >= 0) {
      const __bf16* p = featB + (long long)nidx1 * C_IN + laneHi * 16;
      bf16x8 l0 = *reinterpret_cast<const bf16x8*>(p);
      bf16x8 h0 = *reinterpret_cast<const bf16x8*>(p + 8);
      bf16x8 l1 = *reinterpret_cast<const bf16x8*>(p + 32);
      bf16x8 h1 = *reinterpret_cast<const bf16x8*>(p + 40);
      a10 = __builtin_shufflevector(l0, h0, 0,1,2,3,4,5,6,7,8,9,10,11,12,13,14,15);
      a11 = __builtin_shufflevector(l1, h1, 0,1,2,3,4,5,6,7,8,9,10,11,12,13,14,15);
    }

    // Software-pipelined B: load tile nt+1 while the 4 WMMAs of tile nt execute.
#pragma unroll
    for (int nt = 0; nt < 6; ++nt) {
      const int nx = (nt + 1 < 6) ? nt + 1 : nt;   // last iter reloads (harmless)
      bf16x16 bn0 = *reinterpret_cast<const bf16x16*>(sb + nx * 512);
      bf16x16 bn1 = *reinterpret_cast<const bf16x16*>(sb + (6 + nx) * 512);
      acc0[nt] = __builtin_amdgcn_wmma_f32_16x16x32_bf16(
          false, a00, false, bc0, (short)0, acc0[nt], false, false);
      acc1[nt] = __builtin_amdgcn_wmma_f32_16x16x32_bf16(
          false, a10, false, bc0, (short)0, acc1[nt], false, false);
      acc0[nt] = __builtin_amdgcn_wmma_f32_16x16x32_bf16(
          false, a01, false, bc1, (short)0, acc0[nt], false, false);
      acc1[nt] = __builtin_amdgcn_wmma_f32_16x16x32_bf16(
          false, a11, false, bc1, (short)0, acc1[nt], false, false);
      bc0 = bn0;
      bc1 = bn1;
    }
  }

  // ---- epilogue ----
  float bv[6], gv[6], btv[6];
#pragma unroll
  for (int nt = 0; nt < 6; ++nt) {
    bv[nt]  = bias_p[nt * 16 + n];
    gv[nt]  = gamma_p[nt * 16 + n];
    btv[nt] = beta_p[nt * 16 + n];
  }
  ln_relu_store(acc0, bv, gv, btv, out, mbase + laneHi * 8, n, N);
  ln_relu_store(acc1, bv, gv, btv, out, mbase + 16 + laneHi * 8, n, N);
}

extern "C" void kernel_launch(void* const* d_in, const int* in_sizes, int n_in,
                              void* d_out, int out_size, void* d_ws, size_t ws_size,
                              hipStream_t stream) {
  const float* features = (const float*)d_in[0];
  const int*   coords   = (const int*)d_in[1];
  const float* weight   = (const float*)d_in[2];
  const float* bias     = (const float*)d_in[3];
  const float* gamma_v  = (const float*)d_in[4];
  const float* beta_v   = (const float*)d_in[5];
  float* out = (float*)d_out;
  const int N = in_sizes[0] / C_IN;

  char* ws = (char*)d_ws;
  int* vgrid = (int*)ws;                                   // 160^3 * 4B = 16,384,000
  size_t gridBytes = (size_t)GRIDX * GRIDX * GRIDX * sizeof(int);
  __bf16* featB = (__bf16*)(ws + gridBytes);               // N*64*2B = 33,554,432
  size_t featBytes = (size_t)N * C_IN * sizeof(__bf16);
  __bf16* packW = (__bf16*)(ws + gridBytes + featBytes);   // 27*12*512*2B = 331,776

  const int gcells = GRIDX * GRIDX * GRIDX;
  init_grid_k<<<(gcells + 255) / 256, 256, 0, stream>>>(vgrid, gcells);
  scatter_k<<<(N + 255) / 256, 256, 0, stream>>>(coords, vgrid, N);
  const int n4 = N * C_IN / 4;
  cvt_bf16_k<<<(n4 + 255) / 256, 256, 0, stream>>>(features, featB, n4);
  const int pcount = TAPS * 12 * 512;
  pack_w_k<<<(pcount + 255) / 256, 256, 0, stream>>>(weight, packW);
  subm_conv_ln_relu_k<<<(N + 255) / 256, 256, 0, stream>>>(
      featB, coords, packW, vgrid, bias, gamma_v, beta_v, out, N);
}
